// LlamaAttention_42494406427432
// MI455X (gfx1250) — matrix operand order repaired and checked by reading, not compiled, not run on hardware
//
#include <hip/hip_runtime.h>
#include <hip/hip_bf16.h>

// ---------------- problem constants ----------------
static constexpr int BB  = 2;
static constexpr int SS  = 2048;
static constexpr int HID = 2048;
static constexpr int NHQ = 32;     // query heads
static constexpr int NKV = 8;      // kv heads
static constexpr int DH  = 64;
static constexpr int GRP = NHQ / NKV;
#define ATTN_SCALE 0.125f

typedef __attribute__((ext_vector_type(16))) __bf16 v16bf;
typedef __attribute__((ext_vector_type(8)))  __bf16 v8bf;
typedef __attribute__((ext_vector_type(8)))  float  v8f;

__device__ __forceinline__ __bf16 f2bf(float f) {
  union { float f; unsigned u; } a; a.f = f;
  unsigned u = a.u + 0x7FFFu + ((a.u >> 16) & 1u);   // round-to-nearest-even
  union { unsigned short s; __bf16 b; } r; r.s = (unsigned short)(u >> 16);
  return r.b;
}

__device__ __forceinline__ v16bf mk16(v8bf lo, v8bf hi) {
  return __builtin_shufflevector(lo, hi, 0,1,2,3,4,5,6,7,8,9,10,11,12,13,14,15);
}

__device__ __forceinline__ v8f wmma_bf16(v16bf a, v16bf b, v8f c) {
  // D = A(16x32) * B(32x16) + C, f32 accumulate
  return __builtin_amdgcn_wmma_f32_16x16x32_bf16(false, a, false, b, (short)0, c, false, false);
}

__device__ __forceinline__ v8f vzero8() {
  v8f z = {0.f,0.f,0.f,0.f,0.f,0.f,0.f,0.f}; return z;
}

// Per-lane async global->LDS copy of 16 bytes (GLOBAL_LOAD_ASYNC_TO_LDS_B128,
// GV mode, tracked by ASYNCcnt). ldsAddr = LDS byte address (low 32 bits of
// the generic shared pointer), gAddr = 64-bit global address.
__device__ __forceinline__ void async_ld_b128(unsigned ldsAddr, unsigned long long gAddr) {
  asm volatile("global_load_async_to_lds_b128 %0, %1, off"
               :: "v"(ldsAddr), "v"(gAddr) : "memory");
}
__device__ __forceinline__ void wait_asynccnt0() {
  asm volatile("s_wait_asynccnt 0x0" ::: "memory");
}

// ---------------- fp32 -> bf16 conversion ----------------
__global__ void cvt_f32_bf16(const float* __restrict__ src, __bf16* __restrict__ dst, int n) {
  int i = (blockIdx.x * blockDim.x + threadIdx.x) * 4;
  if (i + 3 < n) {
    float4 v = *(const float4*)(src + i);
    dst[i+0] = f2bf(v.x); dst[i+1] = f2bf(v.y);
    dst[i+2] = f2bf(v.z); dst[i+3] = f2bf(v.w);
  } else {
    for (int j = i; j < n; ++j) dst[j] = f2bf(src[j]);
  }
}

// ---------------- GEMM: C[M,N] = X[M,K] * W[N,K]^T ----------------
// Block = 128 threads (4 waves). Wave tile = 64(M) x 64(N); block = 64 x 256.
// MODE 0: f32 store, row-major [M,N]                  (output projection)
// MODE 1: RoPE + bf16 store into [B, NHp, S, DH]      (Q, K projections)
// MODE 2: bf16 store into [B, NHp, S, DH]             (V projection)
template<int MODE>
__global__ __launch_bounds__(128) void gemm_xwt(
    const __bf16* __restrict__ X, const __bf16* __restrict__ W,
    void* __restrict__ outp,
    const float* __restrict__ cosp, const float* __restrict__ sinp,
    int M, int N, int K, int NHp)
{
  const int lane = threadIdx.x & 31;
  const int wave = threadIdx.x >> 5;
  const int l15  = lane & 15;
  const int hf   = lane >> 4;
  const int m_base = blockIdx.x * 64;
  const int n_base = (blockIdx.y * 4 + wave) * 64;
  if (n_base >= N) return;

  v8f acc[4][4];
  #pragma unroll
  for (int mi = 0; mi < 4; ++mi)
    #pragma unroll
    for (int ni = 0; ni < 4; ++ni) acc[mi][ni] = vzero8();

  #pragma unroll 2
  for (int k0 = 0; k0 < K; k0 += 32) {
    // A fragments (16x32 bf16): lane row = l15; elems 0..7 -> k0+hf*8, 8..15 -> k0+16+hf*8
    v16bf a[4];
    #pragma unroll
    for (int mi = 0; mi < 4; ++mi) {
      const __bf16* xp = X + (size_t)(m_base + mi*16 + l15) * K + k0 + hf*8;
      __builtin_prefetch(xp + 128, 0, 1);            // next k-steps of this row
      a[mi] = mk16(*(const v8bf*)xp, *(const v8bf*)(xp + 16));
    }
    // B fragments (32x16 bf16): lane col = l15; elems 0..7 -> k0+hf*8, 8..15 -> k0+16+hf*8 (row of W)
    #pragma unroll
    for (int ni = 0; ni < 4; ++ni) {
      const __bf16* wp = W + (size_t)(n_base + ni*16 + l15) * K + k0 + hf*8;
      __builtin_prefetch(wp + 128, 0, 1);
      v16bf bfr = mk16(*(const v8bf*)wp, *(const v8bf*)(wp + 16));
      #pragma unroll
      for (int mi = 0; mi < 4; ++mi)
        acc[mi][ni] = wmma_bf16(a[mi], bfr, acc[mi][ni]);
    }
  }

  if (MODE == 0) {
    float* out = (float*)outp;
    #pragma unroll
    for (int mi = 0; mi < 4; ++mi)
      #pragma unroll
      for (int ni = 0; ni < 4; ++ni)
        #pragma unroll
        for (int r = 0; r < 8; ++r) {
          const int m = m_base + mi*16 + 8*hf + r;
          const int n = n_base + ni*16 + l15;
          out[(size_t)m * N + n] = acc[mi][ni][r];
        }
  } else {
    __bf16* out = (__bf16*)outp;
    const int h = n_base / DH;            // wave N-tile == one head (DH == 64)
    if (MODE == 1) {
      // RoPE: q'[d] = q[d]*cos - q[d+32]*sin ; q'[d+32] = q[d+32]*cos + q[d]*sin, d<32
      #pragma unroll
      for (int mi = 0; mi < 4; ++mi)
        #pragma unroll
        for (int nip = 0; nip < 2; ++nip)
          #pragma unroll
          for (int r = 0; r < 8; ++r) {
            const int m = m_base + mi*16 + 8*hf + r;
            const int bb = m / SS, s = m - bb*SS;
            const int d = nip*16 + l15;                 // 0..31
            const size_t cs = ((size_t)bb*SS + s)*DH + d;
            const float cv = cosp[cs], sv = sinp[cs];   // cos/sin equal at d and d+32
            const float lo = acc[mi][nip][r];
            const float hi = acc[mi][nip+2][r];
            const size_t ob = (((size_t)bb*NHp + h)*SS + s)*DH;
            out[ob + d]      = f2bf(lo*cv - hi*sv);
            out[ob + d + 32] = f2bf(hi*cv + lo*sv);
          }
    } else {
      #pragma unroll
      for (int mi = 0; mi < 4; ++mi)
        #pragma unroll
        for (int ni = 0; ni < 4; ++ni)
          #pragma unroll
          for (int r = 0; r < 8; ++r) {
            const int m = m_base + mi*16 + 8*hf + r;
            const int bb = m / SS, s = m - bb*SS;
            const int d = ni*16 + l15;
            out[(((size_t)bb*NHp + h)*SS + s)*DH + d] = f2bf(acc[mi][ni][r]);
          }
    }
  }
}

// ---------------- Flash attention ----------------
// Grid: (S/128, H, B). Block = 128 threads = 4 waves; wave handles 32 q-rows.
// LDS: shared K tile [64k][64d] (filled by async global->LDS),
//      transposed V tile [64d][64k], per-wave P [32][64].
__global__ __launch_bounds__(128) void attn_fa(
    const __bf16* __restrict__ Q,   // [B,H,S,DH]  (RoPE applied)
    const __bf16* __restrict__ Kc,  // [B,HKV,S,DH](RoPE applied)
    const __bf16* __restrict__ Vc,  // [B,HKV,S,DH]
    __bf16* __restrict__ AO)        // [B,S,H*DH]
{
  __shared__ __bf16 ldsK[64*64];
  __shared__ __bf16 ldsVT[64*64];
  __shared__ __bf16 ldsP[4][32*64];

  const int tid  = threadIdx.x;
  const int lane = tid & 31, wave = tid >> 5;
  const int l15  = lane & 15, hf = lane >> 4;
  const int qb0  = blockIdx.x * 128;
  const int h    = blockIdx.y, b = blockIdx.z;
  const int hkv  = h / GRP;
  const int qbase = qb0 + wave * 32;

  const __bf16* Qh = Q  + (((size_t)b*NHQ + h  )*SS) * DH;
  const __bf16* Kh = Kc + (((size_t)b*NKV + hkv)*SS) * DH;
  const __bf16* Vh = Vc + (((size_t)b*NKV + hkv)*SS) * DH;

  // Preload Q A-fragments: [mi(2 x 16 rows)][ks(2 x 32 of DH)]
  v16bf aq[2][2];
  #pragma unroll
  for (int mi = 0; mi < 2; ++mi)
    #pragma unroll
    for (int ks = 0; ks < 2; ++ks) {
      const __bf16* qp = Qh + (size_t)(qbase + mi*16 + l15)*DH + ks*32 + hf*8;
      aq[mi][ks] = mk16(*(const v8bf*)qp, *(const v8bf*)(qp + 16));
    }

  v8f accO[2][4];
  float rm[2][8], rl[2][8];
  #pragma unroll
  for (int mi = 0; mi < 2; ++mi) {
    #pragma unroll
    for (int ni = 0; ni < 4; ++ni) accO[mi][ni] = vzero8();
    #pragma unroll
    for (int r = 0; r < 8; ++r) { rm[mi][r] = -1e30f; rl[mi][r] = 0.f; }
  }

  // Per-thread staging addresses (64 rows x 2 half-rows of 32 elems)
  const int str  = tid >> 1;            // 0..63: tile row
  const int stc0 = (tid & 1) * 32;      // 0 or 32: column base

  const int nkt = blockIdx.x * 2 + 2;   // causal: k-tiles up to block's max q row
  for (int kt = 0; kt < nkt; ++kt) {
    __syncthreads();                    // previous iteration's LDS reads done
    {
      // K tile: async global->LDS (ASYNCcnt path), 4 x 16B per lane
      const __bf16* kp = Kh + (size_t)(kt*64 + str)*DH + stc0;
      const unsigned ldsk = (unsigned)(unsigned long long)(const void*)&ldsK[str*64 + stc0];
      #pragma unroll
      for (int i = 0; i < 4; ++i)
        async_ld_b128(ldsk + i*16, (unsigned long long)(const void*)(kp + i*8));

      // V tile: registers -> transposed LDS (d-major)
      const __bf16* vp = Vh + (size_t)(kt*64 + str)*DH + stc0;
      v8bf vv[4];
      #pragma unroll
      for (int i = 0; i < 4; ++i) vv[i] = *(const v8bf*)(vp + i*8);
      #pragma unroll
      for (int i = 0; i < 4; ++i)
        #pragma unroll
        for (int j = 0; j < 8; ++j)
          ldsVT[(stc0 + i*8 + j)*64 + str] = vv[i][j];

      wait_asynccnt0();                 // K tile landed in LDS
    }
    __syncthreads();

    // S = Q * K^T  (B-frag: col = k-row in tile, elems = d contiguous in ldsK row)
    v8f sc[2][4];
    #pragma unroll
    for (int mi = 0; mi < 2; ++mi)
      #pragma unroll
      for (int ni = 0; ni < 4; ++ni) sc[mi][ni] = vzero8();
    #pragma unroll
    for (int ni = 0; ni < 4; ++ni) {
      const __bf16* b0 = &ldsK[(ni*16 + l15)*64 + hf*8];         // d chunk 0..31
      const __bf16* b1 = b0 + 32;                                 // d chunk 32..63
      v16bf bf0 = mk16(*(const v8bf*)b0, *(const v8bf*)(b0 + 16));
      v16bf bf1 = mk16(*(const v8bf*)b1, *(const v8bf*)(b1 + 16));
      #pragma unroll
      for (int mi = 0; mi < 2; ++mi) {
        sc[mi][ni] = wmma_bf16(aq[mi][0], bf0, sc[mi][ni]);
        sc[mi][ni] = wmma_bf16(aq[mi][1], bf1, sc[mi][ni]);
      }
    }

    // Scale + causal mask + online softmax (row stats duplicated per 16-lane half)
    float al[2][8];
    #pragma unroll
    for (int mi = 0; mi < 2; ++mi)
      #pragma unroll
      for (int r = 0; r < 8; ++r) {
        const int qg = qbase + mi*16 + 8*hf + r;
        float vmax = -1e30f;
        #pragma unroll
        for (int ni = 0; ni < 4; ++ni) {
          const int kg = kt*64 + ni*16 + l15;
          float v = sc[mi][ni][r] * ATTN_SCALE;
          v = (kg > qg) ? -1e30f : v;
          sc[mi][ni][r] = v;
          vmax = fmaxf(vmax, v);
        }
        #pragma unroll
        for (int off = 1; off < 16; off <<= 1)
          vmax = fmaxf(vmax, __shfl_xor(vmax, off, 32));
        const float mo = rm[mi][r];
        const float mn = fmaxf(mo, vmax);
        const float a  = __expf(mo - mn);
        float rs = 0.f;
        #pragma unroll
        for (int ni = 0; ni < 4; ++ni) {
          const float pv = __expf(sc[mi][ni][r] - mn);
          sc[mi][ni][r] = pv;
          rs += pv;
        }
        #pragma unroll
        for (int off = 1; off < 16; off <<= 1)
          rs += __shfl_xor(rs, off, 32);
        rm[mi][r] = mn;
        rl[mi][r] = rl[mi][r] * a + rs;
        al[mi][r] = a;
      }

    #pragma unroll
    for (int mi = 0; mi < 2; ++mi)
      #pragma unroll
      for (int ni = 0; ni < 4; ++ni)
        #pragma unroll
        for (int r = 0; r < 8; ++r)
          accO[mi][ni][r] *= al[mi][r];

    // C-layout -> A-layout via per-wave LDS round-trip
    __bf16* P = ldsP[wave];
    #pragma unroll
    for (int mi = 0; mi < 2; ++mi)
      #pragma unroll
      for (int ni = 0; ni < 4; ++ni)
        #pragma unroll
        for (int r = 0; r < 8; ++r)
          P[(mi*16 + 8*hf + r)*64 + ni*16 + l15] = f2bf(sc[mi][ni][r]);
    asm volatile("s_wait_dscnt 0x0" ::: "memory");   // wave-local LDS ordering

    // O += P * V : preload all P A-fragments, then per-ni load both K-chunk
    // B-fragments before issuing the 4 WMMAs (amortize DScnt waits).
    v16bf ap[2][2];
    #pragma unroll
    for (int mi = 0; mi < 2; ++mi)
      #pragma unroll
      for (int ks = 0; ks < 2; ++ks) {
        const __bf16* pp = &P[(mi*16 + l15)*64 + ks*32 + hf*8];
        ap[mi][ks] = mk16(*(const v8bf*)pp, *(const v8bf*)(pp + 16));
      }
    #pragma unroll
    for (int ni = 0; ni < 4; ++ni) {
      const __bf16* b0 = &ldsVT[(ni*16 + l15)*64 + hf*8];        // k chunk 0..31
      const __bf16* b1 = b0 + 32;                                 // k chunk 32..63
      v16bf bf0 = mk16(*(const v8bf*)b0, *(const v8bf*)(b0 + 16));
      v16bf bf1 = mk16(*(const v8bf*)b1, *(const v8bf*)(b1 + 16));
      #pragma unroll
      for (int mi = 0; mi < 2; ++mi) {
        accO[mi][ni] = wmma_bf16(ap[mi][0], bf0, accO[mi][ni]);
        accO[mi][ni] = wmma_bf16(ap[mi][1], bf1, accO[mi][ni]);
      }
    }
  }

  // Normalize and store into [B,S,H*DH] (row-major for the output GEMM)
  #pragma unroll
  for (int mi = 0; mi < 2; ++mi)
    #pragma unroll
    for (int ni = 0; ni < 4; ++ni)
      #pragma unroll
      for (int r = 0; r < 8; ++r) {
        const int s = qbase + mi*16 + 8*hf + r;
        const int d = ni*16 + l15;
        const float o = accO[mi][ni][r] / rl[mi][r];
        AO[(((size_t)b*SS + s)*NHQ + h)*DH + d] = f2bf(o);
      }
}

// ---------------- host launcher ----------------
extern "C" void kernel_launch(void* const* d_in, const int* in_sizes, int n_in,
                              void* d_out, int out_size, void* d_ws, size_t ws_size,
                              hipStream_t stream) {
  const float* hs   = (const float*)d_in[0];
  const float* cosp = (const float*)d_in[1];
  const float* sinp = (const float*)d_in[2];
  // d_in[3] attention_mask: exactly causal -> computed analytically in-kernel
  const float* Wq   = (const float*)d_in[4];
  const float* Wk   = (const float*)d_in[5];
  const float* Wv   = (const float*)d_in[6];
  const float* Wo   = (const float*)d_in[7];

  char* p = (char*)d_ws;
  auto take = [&](size_t elems) { __bf16* q = (__bf16*)p; p += elems * sizeof(__bf16); return q; };
  __bf16* Xb  = take((size_t)BB*SS*HID);        // 16.8 MB
  __bf16* Wqb = take((size_t)NHQ*DH*HID);       //  8.4 MB
  __bf16* Wkb = take((size_t)NKV*DH*HID);       //  2.1 MB
  __bf16* Wvb = take((size_t)NKV*DH*HID);       //  2.1 MB
  __bf16* Wob = take((size_t)HID*NHQ*DH);       //  8.4 MB
  __bf16* Qb  = take((size_t)BB*NHQ*SS*DH);     // 16.8 MB
  __bf16* Kb  = take((size_t)BB*NKV*SS*DH);     //  4.2 MB
  __bf16* Vb  = take((size_t)BB*NKV*SS*DH);     //  4.2 MB
  __bf16* AOb = take((size_t)BB*SS*NHQ*DH);     // 16.8 MB  (~80 MB total)

  auto cvt = [&](const float* src, __bf16* dst, int n) {
    const int thr = 256, per = 4;
    cvt_f32_bf16<<<(n + thr*per - 1) / (thr*per), thr, 0, stream>>>(src, dst, n);
  };
  cvt(hs, Xb, BB*SS*HID);
  cvt(Wq, Wqb, NHQ*DH*HID);
  cvt(Wk, Wkb, NKV*DH*HID);
  cvt(Wv, Wvb, NKV*DH*HID);
  cvt(Wo, Wob, HID*NHQ*DH);

  const int M = BB * SS;   // 4096
  // Q/K (RoPE fused), V projections
  gemm_xwt<1><<<dim3(M/64, (NHQ*DH)/256), 128, 0, stream>>>(Xb, Wqb, Qb,  cosp, sinp, M, NHQ*DH, HID, NHQ);
  gemm_xwt<1><<<dim3(M/64, (NKV*DH)/256), 128, 0, stream>>>(Xb, Wkb, Kb,  cosp, sinp, M, NKV*DH, HID, NKV);
  gemm_xwt<2><<<dim3(M/64, (NKV*DH)/256), 128, 0, stream>>>(Xb, Wvb, Vb,  cosp, sinp, M, NKV*DH, HID, NKV);

  // Flash attention with GQA
  attn_fa<<<dim3(SS/128, NHQ, BB), 128, 0, stream>>>(Qb, Kb, Vb, AOb);

  // Output projection -> f32 d_out
  gemm_xwt<0><<<dim3(M/64, HID/256), 128, 0, stream>>>(AOb, Wob, d_out, nullptr, nullptr, M, HID, NHQ*DH, 0);
}
